// RAMEmbedding_18691697672527
// MI455X (gfx1250) — 8-bit matrix operand order repaired and checked by reading, compiled, not run on hardware
//
#include <hip/hip_runtime.h>

typedef int v8i  __attribute__((ext_vector_type(8)));
typedef unsigned int u32x4 __attribute__((ext_vector_type(4)));
typedef int i32x4 __attribute__((ext_vector_type(4)));
typedef int i32x8 __attribute__((ext_vector_type(8)));

#define EMBN      1024
#define TABLE     4096
#define NBITS     12
#define BATCH     8
#define SEQ       4096
#define NPOS      (BATCH * SEQ)
#define WPN       128

#define WS_BITS_OFF   0
#define WS_WLO_OFF    (EMBN * WPN * 4)
#define WS_WHI_OFF    (WS_WLO_OFF + EMBN * 16)
#define WS_TOK_OFF    (WS_WHI_OFF + EMBN * 16)

#if defined(__has_builtin)
#if __has_builtin(__builtin_amdgcn_tensor_load_to_lds) && \
    __has_builtin(__builtin_amdgcn_s_wait_tensorcnt)
#define TDM_OK 1
#endif
#endif
#ifndef TDM_OK
#define TDM_OK 0
#endif

__global__ void pack_tables_kernel(const float* __restrict__ tables,
                                   unsigned* __restrict__ bits) {
    int idx = blockIdx.x * blockDim.x + threadIdx.x;
    if (idx >= EMBN * WPN) return;
    const float* src = tables + (size_t)idx * 32;
    unsigned w = 0u;
#pragma unroll
    for (int k = 0; k < 32; ++k)
        w |= (src[k] > 0.5f ? 1u : 0u) << k;
    bits[idx] = w;
}

__global__ void build_w_kernel(const int* __restrict__ conn,
                               unsigned char* __restrict__ wlo,
                               unsigned char* __restrict__ whi) {
    int n = blockIdx.x * blockDim.x + threadIdx.x;
    if (n >= EMBN) return;
    int c[NBITS];
#pragma unroll
    for (int j = 0; j < NBITS; ++j) c[j] = conn[n * NBITS + j];
#pragma unroll
    for (int t = 0; t < 16; ++t) {
        int v = 0;
#pragma unroll
        for (int j = 0; j < NBITS; ++j)
            v += (c[j] == t) ? (1 << (11 - j)) : 0;
        wlo[n * 16 + t] = (unsigned char)(v & 0xFF);
        whi[n * 16 + t] = (unsigned char)((v >> 8) & 0xFF);
    }
}

__global__ void pack_tokens_kernel(const int* __restrict__ tokens,
                                   unsigned* __restrict__ tokmask) {
    int p = blockIdx.x * blockDim.x + threadIdx.x;
    if (p >= NPOS) return;
    unsigned m = 0u;
#pragma unroll
    for (int t = 0; t < NBITS; ++t)
        m |= ((unsigned)tokens[p * NBITS + t] & 1u) << t;
    tokmask[p] = m;
}

__device__ __forceinline__ unsigned spread4(unsigned x) {
    return (x & 1u) | ((x & 2u) << 7) | ((x & 4u) << 14) | ((x & 8u) << 21);
}

__global__ void __launch_bounds__(256)
ram_embed_kernel(const unsigned* __restrict__ tokmask,
                 const unsigned* __restrict__ bits,
                 const unsigned char* __restrict__ wlo,
                 const unsigned char* __restrict__ whi,
                 float* __restrict__ out) {
    __shared__ unsigned lbits[16 * WPN];

    const int g    = blockIdx.x;
    const int tid  = threadIdx.x;
    const int wave = tid >> 5;
    const int lane = tid & 31;
    const int nl   = lane & 15;

#if TDM_OK
    if (wave == 0) {
        const unsigned long long ga =
            (unsigned long long)(const void*)(bits + (size_t)g * 16 * WPN);
        const unsigned lds_byte = (unsigned)(unsigned long long)(const void*)&lbits[0];
        u32x4 g0;
        g0[0] = 1u;
        g0[1] = lds_byte;
        g0[2] = (unsigned)(ga & 0xFFFFFFFFu);
        g0[3] = (unsigned)((ga >> 32) & 0x01FFFFFFu) | (2u << 30);
        i32x8 g1;
        g1[0] = (2 << 16);
        g1[1] = (2048 << 16);
        g1[2] = (1 << 16);
        g1[3] = (int)(2048u << 16);
        g1[4] = 1;
        g1[5] = 2048;
        g1[6] = (int)(2048u << 16);
        g1[7] = 0;
        const i32x4 z4 = {0, 0, 0, 0};
#if __clang_major__ >= 23
        const i32x8 z8 = {0, 0, 0, 0, 0, 0, 0, 0};
        __builtin_amdgcn_tensor_load_to_lds(g0, g1, z4, z4, z8, 0);
#else
        __builtin_amdgcn_tensor_load_to_lds(g0, g1, z4, z4, 0);
#endif
        __builtin_amdgcn_s_wait_tensorcnt(0);
    }
#else
    for (int i = tid; i < 16 * WPN; i += 256)
        lbits[i] = bits[(size_t)g * 16 * WPN + i];
#endif

    const int n = g * 16 + nl;
    const unsigned* wl = (const unsigned*)(wlo + n * 16);
    const unsigned* wh = (const unsigned*)(whi + n * 16);
    const unsigned inlo = (lane < 16) ? 0xFFFFFFFFu : 0u;
    v8i blo = {0, 0, 0, 0, 0, 0, 0, 0};
    v8i bhi = {0, 0, 0, 0, 0, 0, 0, 0};
    blo[0] = (int)wl[(lane >> 4) * 2]; blo[1] = (int)wl[(lane >> 4) * 2 + 1];
    bhi[0] = (int)wh[(lane >> 4) * 2]; bhi[1] = (int)wh[(lane >> 4) * 2 + 1];

    const int posj  = 12 - (n % 13);
    const int mrow  = (lane < 16) ? 0 : 8;
    const int kbase = (lane < 16) ? 0 : 8;

    __syncthreads();

#pragma unroll 2
    for (int i = 0; i < 8; ++i) {
        const int pbase = (blockIdx.y * 64 + wave * 8 + i) * 16;

        const unsigned m = tokmask[pbase + nl];
        v8i a = {0, 0, 0, 0, 0, 0, 0, 0};
        a[0] = (int)spread4(m >> kbase);
        a[1] = (int)spread4(m >> (kbase + 4));

        const v8i c = {0, 0, 0, 0, 0, 0, 0, 0};
        v8i dlo = __builtin_amdgcn_wmma_i32_16x16x64_iu8(
            false, a, false, blo, c, false, false);
        v8i dhi = __builtin_amdgcn_wmma_i32_16x16x64_iu8(
            false, a, false, bhi, c, false, false);

        const int sm = (pbase + mrow) & (SEQ - 1);
        float* po = out + (size_t)(pbase + mrow) * EMBN + n;
#pragma unroll
        for (int r = 0; r < 8; ++r) {
            const int addr = dlo[r] + (dhi[r] << 8);
            const unsigned word = lbits[(nl << 7) + (addr >> 5)];
            const unsigned e  = (word >> (addr & 31)) & 1u;
            const unsigned pb = ((unsigned)(sm + r) >> posj) & 1u;
            __builtin_nontemporal_store((float)(e ^ pb), po + (size_t)r * EMBN);
        }
    }
}

extern "C" void kernel_launch(void* const* d_in, const int* in_sizes, int n_in,
                              void* d_out, int out_size, void* d_ws, size_t ws_size,
                              hipStream_t stream) {
    const int*   tokens = (const int*)d_in[0];
    const float* tables = (const float*)d_in[1];
    const int*   conn   = (const int*)d_in[2];
    float* out = (float*)d_out;

    char* ws = (char*)d_ws;
    unsigned*      bits    = (unsigned*)(ws + WS_BITS_OFF);
    unsigned char* wlo     = (unsigned char*)(ws + WS_WLO_OFF);
    unsigned char* whi     = (unsigned char*)(ws + WS_WHI_OFF);
    unsigned*      tokmask = (unsigned*)(ws + WS_TOK_OFF);

    pack_tables_kernel<<<(EMBN * WPN + 255) / 256, 256, 0, stream>>>(tables, bits);
    build_w_kernel<<<(EMBN + 255) / 256, 256, 0, stream>>>(conn, wlo, whi);
    pack_tokens_kernel<<<(NPOS + 255) / 256, 256, 0, stream>>>(tokens, tokmask);

    dim3 grid(64, 32);
    ram_embed_kernel<<<grid, 256, 0, stream>>>(tokmask, bits, wlo, whi, out);
}
